// PtrNet2_82446192214380
// MI455X (gfx1250) — matrix operand order repaired and checked by reading, not compiled, not run on hardware
//
#include <hip/hip_runtime.h>
#include <cstdint>

// ---------------------------------------------------------------------------
// PtrNet2 forward on gfx1250 (MI455X): f16 WMMA GEMMs fed from LDS via
// GLOBAL_LOAD_ASYNC_TO_LDS_B128 (ASYNCcnt), f32 pointwise everywhere else.
// B = 512, H = 512, CITY = 100, gates = 4H = 2048.
// ---------------------------------------------------------------------------

#define BATCH 512
#define HDIM  512
#define CITY  100
#define GDIM  2048   // 4*H

typedef __attribute__((ext_vector_type(16))) _Float16 v16h;
typedef __attribute__((ext_vector_type(8)))  _Float16 v8h;
typedef __attribute__((ext_vector_type(8)))  float    v8f;

// ---------------------------------------------------------------------------
// Async global -> LDS copy of 16 bytes per lane. LDS offset = low 32 bits of
// the flat pointer (ISA 10.2: LDS aperture addr[31:0] is the LDS offset).
// ---------------------------------------------------------------------------
__device__ __forceinline__ void async_copy16(void* lds_ptr, const void* gptr) {
  unsigned lds = (unsigned)(uintptr_t)lds_ptr;
  unsigned long long ga = (unsigned long long)(uintptr_t)gptr;
  asm volatile("global_load_async_to_lds_b128 %0, %1, off"
               :: "v"(lds), "v"(ga) : "memory");
}
__device__ __forceinline__ void wait_async0() {
  asm volatile("s_wait_asynccnt 0x0" ::: "memory");
}

// ---------------------------------------------------------------------------
// WMMA fragment loaders (v_wmma_f32_16x16x32_f16 layouts, wave32).
// rowp points at the 32 K-contiguous halves for this lane's row/col.
// A: lanes 0-15 halves0..7->K0..7, halves8..15->K16..23; lanes16-31 +8.
// B: same K mapping as A.
// ---------------------------------------------------------------------------
__device__ __forceinline__ v16h load_frag_a(const _Float16* rowp, int hi) {
  const v8h* p0 = reinterpret_cast<const v8h*>(rowp + hi * 8);
  const v8h* p1 = reinterpret_cast<const v8h*>(rowp + 16 + hi * 8);
  v8h lo = *p0, hh = *p1;
  v16h r;
#pragma unroll
  for (int i = 0; i < 8; ++i) { r[i] = lo[i]; r[i + 8] = hh[i]; }
  return r;
}

__device__ __forceinline__ v16h load_frag_b(const _Float16* rowp, int hi) {
  const v8h* p0 = reinterpret_cast<const v8h*>(rowp + hi * 8);
  const v8h* p1 = reinterpret_cast<const v8h*>(rowp + 16 + hi * 8);
  v8h lo = *p0, hh = *p1;
  v16h r;
#pragma unroll
  for (int i = 0; i < 8; ++i) { r[i] = lo[i]; r[i + 8] = hh[i]; }
  return r;
}

// ---------------------------------------------------------------------------
// LDS-staged WMMA GEMM:  out = A1 @ W1^T [+ A2 @ W2^T] [+ bias1 + bias2]
// A: MxK f16 row-major, W: NxK f16 row-major; M, N multiples of 128,
// K multiple of 32. Block = 128x128 tile (8 waves, each 64x32 = 4x2 WMMA
// accumulators); per K-step of 32 the block stages A/B slices into LDS with
// async b128 copies, then every staged fragment feeds 8 WMMAs.
// ---------------------------------------------------------------------------
#define BK 32
#define LDSTRIDE (BK + 8)   // 40 halves = 80 B, keeps 16B alignment, skews banks

__global__ __launch_bounds__(256)
void wmma_gemm_kernel(const _Float16* __restrict__ A1,
                      const _Float16* __restrict__ W1,
                      const _Float16* __restrict__ A2,
                      const _Float16* __restrict__ W2,
                      const float* __restrict__ bias1,
                      const float* __restrict__ bias2,
                      float* __restrict__ outF,
                      _Float16* __restrict__ outH,
                      int M, int N, int K, int relu) {
  __shared__ _Float16 As[128][LDSTRIDE];
  __shared__ _Float16 Bs[128][LDSTRIDE];

  const int nBlocks = N >> 7;
  const int bm = blockIdx.x / nBlocks;
  const int bn = blockIdx.x % nBlocks;
  const int tid  = threadIdx.x;
  const int wid  = tid >> 5;
  const int lane = tid & 31;
  const int row  = lane & 15;       // A row / B col / C col owned by lane
  const int hi   = lane >> 4;
  const int waveM = wid >> 2;       // 0..1 -> 64-row slab
  const int waveN = wid & 3;        // 0..3 -> 32-col slab

  // Staging plan: 128 rows x 32 halves = 512 x 16B transfers per matrix;
  // each of the 256 threads moves transfers {tid, tid+256}.
  const int r0 = tid >> 2, s0 = tid & 3;   // row, 8-half segment
  const int r1 = r0 + 64;
  _Float16* lA0 = &As[r0][s0 * 8];
  _Float16* lA1 = &As[r1][s0 * 8];
  _Float16* lB0 = &Bs[r0][s0 * 8];
  _Float16* lB1 = &Bs[r1][s0 * 8];

  v8f acc[4][2];
#pragma unroll
  for (int mt = 0; mt < 4; ++mt)
#pragma unroll
    for (int nt = 0; nt < 2; ++nt) acc[mt][nt] = {};

  const _Float16* Ap[2] = {A1, A2};
  const _Float16* Wp[2] = {W1, W2};
  const int npass = (A2 != nullptr) ? 2 : 1;

  for (int p = 0; p < npass; ++p) {
    const _Float16* Ag = Ap[p] + (size_t)bm * 128 * K;
    const _Float16* Wg = Wp[p] + (size_t)bn * 128 * K;
    for (int k0 = 0; k0 < K; k0 += BK) {
      __syncthreads();   // previous K-step compute done before overwrite
      async_copy16(lA0, Ag + (size_t)r0 * K + k0 + s0 * 8);
      async_copy16(lA1, Ag + (size_t)r1 * K + k0 + s0 * 8);
      async_copy16(lB0, Wg + (size_t)r0 * K + k0 + s0 * 8);
      async_copy16(lB1, Wg + (size_t)r1 * K + k0 + s0 * 8);
      wait_async0();
      __syncthreads();   // all waves' staged data visible

      v16h bf0 = load_frag_b(&Bs[waveN * 32 + row][0], hi);
      v16h bf1 = load_frag_b(&Bs[waveN * 32 + 16 + row][0], hi);
#pragma unroll
      for (int mt = 0; mt < 4; ++mt) {
        v16h af = load_frag_a(&As[waveM * 64 + mt * 16 + row][0], hi);
        acc[mt][0] = __builtin_amdgcn_wmma_f32_16x16x32_f16(
            false, af, false, bf0, (short)0, acc[mt][0], false, false);
        acc[mt][1] = __builtin_amdgcn_wmma_f32_16x16x32_f16(
            false, af, false, bf1, (short)0, acc[mt][1], false, false);
      }
    }
  }

#pragma unroll
  for (int nt = 0; nt < 2; ++nt) {
    const int n = bn * 128 + waveN * 32 + nt * 16 + row;
    float badd = 0.0f;
    if (bias1) badd += bias1[n];
    if (bias2) badd += bias2[n];
#pragma unroll
    for (int mt = 0; mt < 4; ++mt) {
#pragma unroll
      for (int r = 0; r < 8; ++r) {
        const int m = bm * 128 + waveM * 64 + mt * 16 + r + hi * 8;
        float val = acc[mt][nt][r] + badd;
        if (relu) val = fmaxf(val, 0.0f);
        if (outF) outF[(size_t)m * N + n] = val;
        if (outH) outH[(size_t)m * N + n] = (_Float16)val;
      }
    }
  }
}

// ---------------------------------------------------------------------------
// LSTM pointwise: gates (B x 4H, order i,f,g,o) -> update h, c; emit f16 h.
// ---------------------------------------------------------------------------
__global__ void lstm_point_kernel(const float* __restrict__ gates,
                                  float* __restrict__ h,
                                  float* __restrict__ c,
                                  _Float16* __restrict__ h16,
                                  _Float16* __restrict__ enc_store,
                                  long enc_rowstride) {
  const int idx = blockIdx.x * blockDim.x + threadIdx.x;   // 0 .. B*H-1
  const int b = idx >> 9;
  const int j = idx & (HDIM - 1);
  const float* g = gates + (size_t)b * GDIM;
  const float gi = g[j];
  const float gf = g[HDIM + j];
  const float gg = g[2 * HDIM + j];
  const float go = g[3 * HDIM + j];
  const float si = 1.0f / (1.0f + __expf(-gi));
  const float sf = 1.0f / (1.0f + __expf(-gf));
  const float so = 1.0f / (1.0f + __expf(-go));
  const float cn = sf * c[idx] + si * tanhf(gg);
  const float hn = so * tanhf(cn);
  c[idx] = cn;
  h[idx] = hn;
  h16[idx] = (_Float16)hn;
  if (enc_store) enc_store[(size_t)b * enc_rowstride + j] = (_Float16)hn;
}

// emb_t[b][j] = x[b][t][0]*Wemb[j][0] + x[b][t][1]*Wemb[j][1]   (f16 out)
__global__ void embed_kernel(const float* __restrict__ x,
                             const float* __restrict__ Wemb,
                             _Float16* __restrict__ emb, int t) {
  const int idx = blockIdx.x * blockDim.x + threadIdx.x;   // B*H
  const int b = idx >> 9;
  const int j = idx & (HDIM - 1);
  const float x0 = x[(size_t)b * CITY * 2 + t * 2 + 0];
  const float x1 = x[(size_t)b * CITY * 2 + t * 2 + 1];
  emb[idx] = (_Float16)(x0 * Wemb[2 * j] + x1 * Wemb[2 * j + 1]);
}

__global__ void f32_to_f16_kernel(const float* __restrict__ in,
                                  _Float16* __restrict__ out, int n) {
  const int i = blockIdx.x * blockDim.x + threadIdx.x;
  if (i < n) out[i] = (_Float16)in[i];
}

__global__ void zero_buffers_kernel(float* __restrict__ h, float* __restrict__ c,
                                    _Float16* __restrict__ h16, int n) {
  const int i = blockIdx.x * blockDim.x + threadIdx.x;
  if (i < n) { h[i] = 0.0f; c[i] = 0.0f; h16[i] = (_Float16)0.0f; }
}

// ---------------------------------------------------------------------------
// Attention: u[c] = v . tanh(u1[b,c,:] + u2[b,:]); a = softmax(u);
//            dec_in[b,:] = f16( sum_c a[c] * enc_h[b,c,:] )
// One workgroup (256 threads = 8 waves) per batch row.
// ---------------------------------------------------------------------------
__global__ void attention_kernel(const _Float16* __restrict__ u1,
                                 const float* __restrict__ u2,
                                 const _Float16* __restrict__ ench,
                                 const float* __restrict__ v,
                                 _Float16* __restrict__ dec_in) {
  __shared__ float su[CITY];
  __shared__ float u2s[HDIM];
  const int b = blockIdx.x;
  const int tid = threadIdx.x;
  const int wid = tid >> 5;
  const int lane = tid & 31;

  for (int hh = tid; hh < HDIM; hh += 256) u2s[hh] = u2[(size_t)b * HDIM + hh];
  __syncthreads();

  const _Float16* u1b = u1 + (size_t)b * CITY * HDIM;
  for (int cc = wid; cc < CITY; cc += 8) {       // 8 waves split the cities
    const _Float16* u1c = u1b + (size_t)cc * HDIM;
    float s = 0.0f;
    for (int hh = lane; hh < HDIM; hh += 32)
      s += v[hh] * tanhf((float)u1c[hh] + u2s[hh]);
#pragma unroll
    for (int off = 16; off > 0; off >>= 1) s += __shfl_xor(s, off, 32);
    if (lane == 0) su[cc] = s;
  }
  __syncthreads();

  if (tid == 0) {                                 // softmax over 100 entries
    float mx = su[0];
    for (int cc = 1; cc < CITY; ++cc) mx = fmaxf(mx, su[cc]);
    float sum = 0.0f;
    for (int cc = 0; cc < CITY; ++cc) { float e = __expf(su[cc] - mx); su[cc] = e; sum += e; }
    const float inv = 1.0f / sum;
    for (int cc = 0; cc < CITY; ++cc) su[cc] *= inv;
  }
  __syncthreads();

  const _Float16* eb = ench + (size_t)b * CITY * HDIM;
  for (int hh = tid; hh < HDIM; hh += 256) {
    float acc = 0.0f;
    for (int cc = 0; cc < CITY; ++cc)
      acc += su[cc] * (float)eb[(size_t)cc * HDIM + hh];
    dec_in[(size_t)b * HDIM + hh] = (_Float16)acc;
  }
}

// out[b] = sum_h t1[b][h] * fc2[h]
__global__ void final_dot_kernel(const _Float16* __restrict__ t1,
                                 const float* __restrict__ fc2,
                                 float* __restrict__ out) {
  __shared__ float red[256];
  const int b = blockIdx.x;
  const int tid = threadIdx.x;
  float s = 0.0f;
  for (int hh = tid; hh < HDIM; hh += 256)
    s += (float)t1[(size_t)b * HDIM + hh] * fc2[hh];
  red[tid] = s;
  __syncthreads();
  for (int st = 128; st > 0; st >>= 1) {
    if (tid < st) red[tid] += red[tid + st];
    __syncthreads();
  }
  if (tid == 0) out[b] = red[0];
}

// ---------------------------------------------------------------------------
extern "C" void kernel_launch(void* const* d_in, const int* in_sizes, int n_in,
                              void* d_out, int out_size, void* d_ws, size_t ws_size,
                              hipStream_t stream) {
  (void)in_sizes; (void)n_in; (void)out_size; (void)ws_size;

  const float* x        = (const float*)d_in[0];   // (B, CITY, 2)
  const float* dec_i1   = (const float*)d_in[1];   // (B, H)
  const float* W_emb    = (const float*)d_in[2];   // (H, 2)
  const float* enc_Wih  = (const float*)d_in[3];   // (4H, H)
  const float* enc_Whh  = (const float*)d_in[4];
  const float* enc_bih  = (const float*)d_in[5];
  const float* enc_bhh  = (const float*)d_in[6];
  const float* dec_Wih  = (const float*)d_in[7];
  const float* dec_Whh  = (const float*)d_in[8];
  const float* dec_bih  = (const float*)d_in[9];
  const float* dec_bhh  = (const float*)d_in[10];
  const float* W_q      = (const float*)d_in[11];
  const float* W_ref    = (const float*)d_in[12];
  const float* v        = (const float*)d_in[13];  // (1, H)
  const float* fc1      = (const float*)d_in[14];
  const float* fc2      = (const float*)d_in[15];
  float* out            = (float*)d_out;           // (B,)

  // -------- workspace layout (all 256B aligned, ~123 MB total) -------------
  char* ws = (char*)d_ws;
  size_t off = 0;
  auto alloc = [&](size_t bytes) -> char* {
    char* p = ws + off;
    off += (bytes + 255) & ~(size_t)255;
    return p;
  };
  _Float16* Wih_e  = (_Float16*)alloc((size_t)GDIM * HDIM * 2);
  _Float16* Whh_e  = (_Float16*)alloc((size_t)GDIM * HDIM * 2);
  _Float16* Wih_d  = (_Float16*)alloc((size_t)GDIM * HDIM * 2);
  _Float16* Whh_d  = (_Float16*)alloc((size_t)GDIM * HDIM * 2);
  _Float16* Wq_h   = (_Float16*)alloc((size_t)HDIM * HDIM * 2);
  _Float16* Wref_h = (_Float16*)alloc((size_t)HDIM * HDIM * 2);
  _Float16* Wfc1_h = (_Float16*)alloc((size_t)HDIM * HDIM * 2);
  _Float16* ench   = (_Float16*)alloc((size_t)BATCH * CITY * HDIM * 2); // 52 MB
  _Float16* u1     = (_Float16*)alloc((size_t)BATCH * CITY * HDIM * 2); // 52 MB
  float*    gates  = (float*)   alloc((size_t)BATCH * GDIM * 4);        // 4 MB
  _Float16* emb_t  = (_Float16*)alloc((size_t)BATCH * HDIM * 2);
  float*    hbuf   = (float*)   alloc((size_t)BATCH * HDIM * 4);
  float*    cbuf   = (float*)   alloc((size_t)BATCH * HDIM * 4);
  _Float16* h16    = (_Float16*)alloc((size_t)BATCH * HDIM * 2);
  _Float16* dec_in = (_Float16*)alloc((size_t)BATCH * HDIM * 2);
  float*    u2     = (float*)   alloc((size_t)BATCH * HDIM * 4);
  _Float16* t1h    = (_Float16*)alloc((size_t)BATCH * HDIM * 2);

  const int BH = BATCH * HDIM;        // 262144
  const int bhBlocks = BH / 256;      // 1024

  // -------- one-time weight conversions (f32 -> f16) -----------------------
  {
    const int nw = GDIM * HDIM;
    f32_to_f16_kernel<<<(nw + 255) / 256, 256, 0, stream>>>(enc_Wih, Wih_e, nw);
    f32_to_f16_kernel<<<(nw + 255) / 256, 256, 0, stream>>>(enc_Whh, Whh_e, nw);
    f32_to_f16_kernel<<<(nw + 255) / 256, 256, 0, stream>>>(dec_Wih, Wih_d, nw);
    f32_to_f16_kernel<<<(nw + 255) / 256, 256, 0, stream>>>(dec_Whh, Whh_d, nw);
    const int ns = HDIM * HDIM;
    f32_to_f16_kernel<<<(ns + 255) / 256, 256, 0, stream>>>(W_q,   Wq_h,   ns);
    f32_to_f16_kernel<<<(ns + 255) / 256, 256, 0, stream>>>(W_ref, Wref_h, ns);
    f32_to_f16_kernel<<<(ns + 255) / 256, 256, 0, stream>>>(fc1,   Wfc1_h, ns);
  }
  zero_buffers_kernel<<<bhBlocks, 256, 0, stream>>>(hbuf, cbuf, h16, BH);

  // -------- encoder: 100 LSTM steps ----------------------------------------
  // gates GEMM: M=512, N=2048 -> (512/128)*(2048/128) = 64 blocks
  for (int t = 0; t < CITY; ++t) {
    embed_kernel<<<bhBlocks, 256, 0, stream>>>(x, W_emb, emb_t, t);
    wmma_gemm_kernel<<<64, 256, 0, stream>>>(emb_t, Wih_e, h16, Whh_e,
                                             enc_bih, enc_bhh, gates, nullptr,
                                             BATCH, GDIM, HDIM, 0);
    lstm_point_kernel<<<bhBlocks, 256, 0, stream>>>(gates, hbuf, cbuf, h16,
                                                    ench + (size_t)t * HDIM,
                                                    (long)CITY * HDIM);
  }

  // -------- u1 = enc_h @ W_ref^T : M=51200, N=512 -> 400*4 = 1600 blocks ---
  wmma_gemm_kernel<<<1600, 256, 0, stream>>>(ench, Wref_h, nullptr, nullptr,
                                             nullptr, nullptr, nullptr, u1,
                                             BATCH * CITY, HDIM, HDIM, 0);

  // -------- decoder: 100 steps of LSTM + attention --------------------------
  f32_to_f16_kernel<<<bhBlocks, 256, 0, stream>>>(dec_i1, dec_in, BH);
  for (int t = 0; t < CITY; ++t) {
    wmma_gemm_kernel<<<64, 256, 0, stream>>>(dec_in, Wih_d, h16, Whh_d,
                                             dec_bih, dec_bhh, gates, nullptr,
                                             BATCH, GDIM, HDIM, 0);
    lstm_point_kernel<<<bhBlocks, 256, 0, stream>>>(gates, hbuf, cbuf, h16,
                                                    nullptr, 0);
    if (t < CITY - 1) {  // last step's glimpse is discarded by the reference
      wmma_gemm_kernel<<<16, 256, 0, stream>>>(h16, Wq_h, nullptr, nullptr,
                                               nullptr, nullptr, u2, nullptr,
                                               BATCH, HDIM, HDIM, 0);
      attention_kernel<<<BATCH, 256, 0, stream>>>(u1, u2, ench, v, dec_in);
    }
  }

  // -------- head: relu(h_last @ fc1^T) @ fc2^T ------------------------------
  wmma_gemm_kernel<<<16, 256, 0, stream>>>(h16, Wfc1_h, nullptr, nullptr,
                                           nullptr, nullptr, nullptr, t1h,
                                           BATCH, HDIM, HDIM, 1);
  final_dot_kernel<<<BATCH, 256, 0, stream>>>(t1h, fc2, out);
}
